// ResAttBlock_7361573945893
// MI455X (gfx1250) — matrix operand order repaired and checked by reading, not compiled, not run on hardware
//
#include <hip/hip_runtime.h>
#include <hip/hip_bf16.h>

// ---------------------------------------------------------------------------
// CDNA5 / gfx1250 implementation of ResBlock + SFT + SpatialSelfAttention.
// All matmuls (implicit-GEMM convs, QK^T, P*V) use v_wmma_f32_16x16x32_f16.
// Attention is flash-style; K/V/Q tiles staged into LDS via the Tensor Data
// Mover (TENSOR_LOAD_TO_LDS + s_wait_tensorcnt) when the builtin is present.
// ---------------------------------------------------------------------------

typedef __attribute__((ext_vector_type(16))) _Float16 v16h;
typedef __attribute__((ext_vector_type(8)))  float    v8f;
typedef __attribute__((ext_vector_type(4)))  unsigned int v4u;
typedef __attribute__((ext_vector_type(8)))  int      v8i;
typedef __attribute__((ext_vector_type(4)))  int      v4i;

#define WMMA_F32_F16(a, b, c) \
    __builtin_amdgcn_wmma_f32_16x16x32_f16(false, (a), false, (b), (short)0, (c), false, false)

// ---- TDM availability / arity detection -----------------------------------
#if __has_builtin(__builtin_amdgcn_tensor_load_to_lds)
#  if __has_include(<hip/amd_detail/amd_gfx1250_TDM.h>)
#    define ATT_TDM_6ARG 1
#  else
#    define ATT_TDM_5ARG 1
#  endif
#endif

#if defined(ATT_TDM_5ARG) || defined(ATT_TDM_6ARG)
// 2D tile load: tensor is [tensor_d1 rows][tensor_d0 elems], row stride
// stride_elems (f16 elements, data_size = 2 bytes).  Tile = tile_d1 x tile_d0.
// D# packing per CDNA5 ISA 8.3/8.4 (group0: count/lds/global/type, group1:
// data_size, dims, tile dims, dim0 stride).
__device__ __forceinline__ void tdm_load_2d_f16(unsigned int lds_off, const void* gptr,
                                                unsigned int tensor_d0, unsigned int tensor_d1,
                                                unsigned int tile_d0, unsigned int tile_d1,
                                                unsigned int stride_elems) {
    unsigned long long ga = (unsigned long long)(uintptr_t)gptr;
    v4u g0;
    g0.x = 1u;                                             // count=1, user desc
    g0.y = lds_off;                                        // lds_addr (bytes)
    g0.z = (unsigned int)ga;                               // global_addr[31:0]
    g0.w = (unsigned int)((ga >> 32) & 0x1FFFFFFu) | (2u << 30); // [56:32] | type=2
    v8i g1;
    g1[0] = (int)(1u << 16);                               // data_size=1 -> 2 bytes
    g1[1] = (int)((tensor_d0 & 0xFFFFu) << 16);            // tensor_dim0[15:0] @ [63:48]
    g1[2] = (int)(((tensor_d0 >> 16) & 0xFFFFu) |          // tensor_dim0[31:16]
                  ((tensor_d1 & 0xFFFFu) << 16));          // tensor_dim1[15:0]
    g1[3] = (int)(((tensor_d1 >> 16) & 0xFFFFu) |          // tensor_dim1[31:16]
                  ((tile_d0 & 0xFFFFu) << 16));            // tile_dim0
    g1[4] = (int)(tile_d1 & 0xFFFFu);                      // tile_dim1, tile_dim2=0
    g1[5] = (int)stride_elems;                             // tensor_dim0_stride[31:0]
    g1[6] = 0;
    g1[7] = 0;
    v4i z4 = {0, 0, 0, 0};
#if defined(ATT_TDM_6ARG)
    v8i z8 = {0, 0, 0, 0, 0, 0, 0, 0};
    __builtin_amdgcn_tensor_load_to_lds(g0, g1, z4, z4, z8, 0);
#else
    __builtin_amdgcn_tensor_load_to_lds(g0, g1, z4, z4, 0);
#endif
}
#endif

// ---- WMMA fragment loaders (ISA 7.12.2 16-bit layouts) ---------------------
// A matrix 16x32 f16: lane L holds row M=L&15; lo half-lanes K={0..7,16..23},
// hi half-lanes K={8..15,24..31}, packed as K-pairs per dword.
__device__ __forceinline__ v16h frag_a16x32(const _Float16* sm, int ld, int koff) {
    int lane = threadIdx.x & 31;
    int hi   = lane >> 4;
    const uint32_t* p = (const uint32_t*)(sm + (size_t)(lane & 15) * ld + koff);
    union { v16h v; uint32_t u[8]; } f;
#pragma unroll
    for (int i = 0; i < 8; ++i) {
        int d = (i & 3) + ((i >> 2) << 3) + (hi << 2);  // dword index in row
        f.u[i] = p[d];
    }
    return f.v;
}

// B matrix 32x16 f16 staged transposed in LDS as Bt[n][k] (k contiguous):
// lane L holds column N=L&15; lo half-lanes K={0..7,16..23}, hi half-lanes K={8..15,24..31}.
__device__ __forceinline__ v16h frag_b32x16(const _Float16* smT, int ld, int koff) {
    int lane = threadIdx.x & 31;
    int hi   = lane >> 4;
    const uint32_t* p = (const uint32_t*)(smT + (size_t)(lane & 15) * ld + koff);
    union { v16h v; uint32_t u[8]; } f;
#pragma unroll
    for (int i = 0; i < 8; ++i) f.u[i] = p[(i & 3) + ((i >> 2) << 3) + (hi << 2)];
    return f.v;
}

// ---------------------------------------------------------------------------
// Fused GroupNorm (+ optional PReLU), f32 in -> f16 out.
// One block per (batch, group). 256 threads.
// ---------------------------------------------------------------------------
__global__ void gn_prelu_kernel(const float* __restrict__ in, _Float16* __restrict__ out,
                                const float* __restrict__ g, const float* __restrict__ b,
                                const float* __restrict__ a, // nullable scalar PReLU slope
                                int C, int HW, int groups) {
    int batch = blockIdx.x / groups;
    int grp   = blockIdx.x % groups;
    int cg    = C / groups;
    size_t base = ((size_t)batch * C + (size_t)grp * cg) * HW;
    int n = cg * HW;

    float s = 0.f, ss = 0.f;
    for (int i = threadIdx.x; i < n; i += blockDim.x) {
        float x = in[base + i];
        s += x; ss += x * x;
    }
    __shared__ float rs[256], rss[256];
    rs[threadIdx.x] = s; rss[threadIdx.x] = ss;
    __syncthreads();
    for (int st = 128; st > 0; st >>= 1) {
        if ((int)threadIdx.x < st) {
            rs[threadIdx.x]  += rs[threadIdx.x + st];
            rss[threadIdx.x] += rss[threadIdx.x + st];
        }
        __syncthreads();
    }
    float mean = rs[0] / (float)n;
    float var  = rss[0] / (float)n - mean * mean;
    float inv  = rsqrtf(var + 1e-6f);
    bool  use_prelu = (a != nullptr);
    float alpha = use_prelu ? a[0] : 0.f;

    for (int i = threadIdx.x; i < n; i += blockDim.x) {
        int c = grp * cg + i / HW;
        float x = (in[base + i] - mean) * inv * g[c] + b[c];
        if (use_prelu) x = (x >= 0.f) ? x : alpha * x;
        out[base + i] = (_Float16)x;
    }
}

// ---------------------------------------------------------------------------
// Implicit-GEMM conv via WMMA.  M = 16 Cout, N = 128 pixels (4 waves x 32),
// K = Cin*ks*ks stepped by 32 through LDS.  2 WMMAs/wave/K-step, A reused.
// grid = (pixTiles, Cout/16, B), block = 128.
// ---------------------------------------------------------------------------
__global__ void conv_wmma_kernel(const _Float16* __restrict__ in,
                                 const float* __restrict__ w,
                                 const float* __restrict__ bias,
                                 float* __restrict__ out32,        // nullable
                                 _Float16* __restrict__ out16,     // nullable [b][c][px]
                                 _Float16* __restrict__ out16T,    // nullable [b][px][c]
                                 const float* __restrict__ res,    // nullable residual add
                                 int Cin, int Cout,
                                 int Hin, int Win, int Hout, int Wout,
                                 int ks, int stride, int pad, int act) {
    __shared__ _Float16 As[16 * 32];    // [cout_row][k]
    __shared__ _Float16 Bs[128 * 32];   // [pixel][k] (transposed for frag_b)

    const int tid   = threadIdx.x;
    const int lane  = tid & 31;
    const int wave  = tid >> 5;
    const int ptile = blockIdx.x;
    const int ct    = blockIdx.y;
    const int bb    = blockIdx.z;

    const int rs2    = ks * ks;
    const int K      = Cin * rs2;
    const int Ksteps = K >> 5;   // K is a multiple of 32 for all our shapes
    const size_t wbase = (size_t)ct * 16 * K;

    v8f acc0, acc1;
#pragma unroll
    for (int r = 0; r < 8; ++r) { acc0[r] = 0.f; acc1[r] = 0.f; }

    for (int kk = 0; kk < Ksteps; ++kk) {
        const int kbase = kk << 5;
        // stage A: 16x32 weights (f32 -> f16)
        for (int idx = tid; idx < 16 * 32; idx += 128) {
            int row = idx >> 5, kc = idx & 31;
            As[idx] = (_Float16)w[wbase + (size_t)row * K + kbase + kc];
        }
        // prefetch next weight slab (emits global_prefetch_b8)
        if (kk + 1 < Ksteps)
            __builtin_prefetch(&w[wbase + kbase + 32], 0, 0);
        // stage B: 32 k-values x 128 pixels, stored [pixel][k]
        for (int idx = tid; idx < 128 * 32; idx += 128) {
            int n  = idx >> 5, kc = idx & 31;
            int kg = kbase + kc;
            int cin = kg / rs2, off = kg % rs2;
            int dy = off / ks, dx = off % ks;
            int pglob = ptile * 128 + n;
            int oy = pglob / Wout, ox = pglob % Wout;
            int iy = oy * stride + dy - pad;
            int ix = ox * stride + dx - pad;
            _Float16 val = (_Float16)0.f;
            if (iy >= 0 && iy < Hin && ix >= 0 && ix < Win)
                val = in[((size_t)bb * Cin + cin) * Hin * Win + (size_t)iy * Win + ix];
            Bs[(size_t)n * 32 + kc] = val;
        }
        __syncthreads();

        v16h af  = frag_a16x32(As, 32, 0);
        v16h bf0 = frag_b32x16(Bs + (size_t)(wave * 32) * 32, 32, 0);
        v16h bf1 = frag_b32x16(Bs + (size_t)(wave * 32 + 16) * 32, 32, 0);
        acc0 = WMMA_F32_F16(af, bf0, acc0);
        acc1 = WMMA_F32_F16(af, bf1, acc1);
        __syncthreads();
    }

    // D layout: VGPR r -> row M = r + 8*hi ; col N = lane&15
    const int hi = lane >> 4;
    const int nn = lane & 15;
    const size_t PT = (size_t)Hout * Wout;
#pragma unroll
    for (int half = 0; half < 2; ++half) {
        const v8f& acc = half ? acc1 : acc0;
        int px = ptile * 128 + wave * 32 + half * 16 + nn;
#pragma unroll
        for (int r = 0; r < 8; ++r) {
            int m = r + (hi << 3);
            int co = ct * 16 + m;
            float v = acc[r] + bias[co];
            if (act == 1) v = (v >= 0.f) ? v : 0.2f * v;
            size_t o = ((size_t)bb * Cout + co) * PT + px;
            if (res)    v += res[o];
            if (out32)  out32[o] = v;
            if (out16)  out16[o] = (_Float16)v;
            if (out16T) out16T[((size_t)bb * PT + px) * Cout + co] = (_Float16)v;
        }
    }
}

// ---------------------------------------------------------------------------
// Flash attention: one wave per 16-query tile.
//   qT,kT f16 token-major [B][HW][C];  v f16 channel-major [B][C][HW].
// K/V/Q tiles staged to LDS via TDM when available, else manual.
// grid = (HW/16, B), block = 32.
// ---------------------------------------------------------------------------
__global__ void attn_flash_kernel(const _Float16* __restrict__ qT,
                                  const _Float16* __restrict__ kT,
                                  const _Float16* __restrict__ v,
                                  _Float16* __restrict__ out,   // [B][C][HW]
                                  int C, int HW, float scale) {
    __shared__ _Float16 qa[16 * 128];  // [i][c]   A for S-WMMA
    __shared__ _Float16 kt[32 * 128];  // [j][c]   Bt for S-WMMA (n=j rows)
    __shared__ _Float16 vt[128 * 32];  // [c][j]   Bt for O-WMMA (n=c rows)
    __shared__ _Float16 ps[16 * 32];   // [i][j]   A for O-WMMA

    const int lane = threadIdx.x;
    const int b    = blockIdx.y;
    const int i0   = blockIdx.x * 16;
    const size_t tb = (size_t)b * HW * C;   // base for token-major qT/kT
    const size_t cb = (size_t)b * C * HW;   // base for channel-major v / out
    const int hi = lane >> 4;
    const int nl = lane & 15;

#if defined(ATT_TDM_5ARG) || defined(ATT_TDM_6ARG)
    const unsigned int qa_off = (unsigned int)(uintptr_t)&qa[0];
    const unsigned int kt_off = (unsigned int)(uintptr_t)&kt[0];
    const unsigned int vt_off = (unsigned int)(uintptr_t)&vt[0];
    // Q tile: 16 rows of C from qT (row stride C, contiguous rows)
    tdm_load_2d_f16(qa_off, qT + tb + (size_t)i0 * C, C, HW, C, 16, C);
#else
    for (int idx = lane; idx < 16 * 128; idx += 32) {
        int i = idx >> 7, c = idx & 127;
        qa[idx] = qT[tb + (size_t)(i0 + i) * C + c];
    }
#endif

    v8f accO[8];
#pragma unroll
    for (int t = 0; t < 8; ++t)
#pragma unroll
        for (int r = 0; r < 8; ++r) accO[t][r] = 0.f;
    float rowM[8], rowL[8];
#pragma unroll
    for (int r = 0; r < 8; ++r) { rowM[r] = -1e30f; rowL[r] = 0.f; }

    for (int j0 = 0; j0 < HW; j0 += 32) {
        __syncthreads();
#if defined(ATT_TDM_5ARG) || defined(ATT_TDM_6ARG)
        // K tile: 32 token rows x C   -> kt[j][c]
        tdm_load_2d_f16(kt_off, kT + tb + (size_t)j0 * C, C, HW, C, 32, C);
        // V tile: 128 channel rows x 32 tokens -> vt[c][j]
        tdm_load_2d_f16(vt_off, v + cb + j0, HW, C, 32, 128, HW);
        __builtin_amdgcn_s_wait_tensorcnt(0);
#else
        for (int idx = lane; idx < 32 * 128; idx += 32) {
            int j = idx >> 7, c = idx & 127;
            kt[(size_t)j * 128 + c] = kT[tb + (size_t)(j0 + j) * C + c];
            vt[(size_t)c * 32 + j]  = v[cb + (size_t)c * HW + j0 + j];
        }
#endif
        __syncthreads();

        // S = Q K^T : 16 x 32 scores via two accumulators, K-dim = C = 128
        v8f s0, s1;
#pragma unroll
        for (int r = 0; r < 8; ++r) { s0[r] = 0.f; s1[r] = 0.f; }
#pragma unroll
        for (int kk = 0; kk < 4; ++kk) {
            v16h af = frag_a16x32(qa, 128, kk * 32);
            v16h b0 = frag_b32x16(kt,            128, kk * 32);  // j cols 0..15
            v16h b1 = frag_b32x16(kt + 16 * 128, 128, kk * 32);  // j cols 16..31
            s0 = WMMA_F32_F16(af, b0, s0);
            s1 = WMMA_F32_F16(af, b1, s1);
        }

        // Online softmax (row m = r + 8*hi; 16 cols spread across 16 lanes)
        float p0[8], p1[8], tmax[8];
#pragma unroll
        for (int r = 0; r < 8; ++r) {
            p0[r] = s0[r] * scale;
            p1[r] = s1[r] * scale;
            tmax[r] = fmaxf(p0[r], p1[r]);
        }
#pragma unroll
        for (int r = 0; r < 8; ++r)
            for (int m = 1; m < 16; m <<= 1)
                tmax[r] = fmaxf(tmax[r], __shfl_xor(tmax[r], m, 16));
        float alpha[8];
#pragma unroll
        for (int r = 0; r < 8; ++r) {
            float nm = fmaxf(rowM[r], tmax[r]);
            alpha[r] = __expf(rowM[r] - nm);
            rowM[r]  = nm;
            p0[r] = __expf(p0[r] - nm);
            p1[r] = __expf(p1[r] - nm);
        }
#pragma unroll
        for (int r = 0; r < 8; ++r) {
            float psum = p0[r] + p1[r];
            for (int m = 1; m < 16; m <<= 1)
                psum += __shfl_xor(psum, m, 16);
            rowL[r] = rowL[r] * alpha[r] + psum;
        }
#pragma unroll
        for (int t = 0; t < 8; ++t)
#pragma unroll
            for (int r = 0; r < 8; ++r) accO[t][r] *= alpha[r];

        // D-layout -> A-layout via LDS for P
#pragma unroll
        for (int r = 0; r < 8; ++r) {
            int m = r + (hi << 3);
            ps[m * 32 + nl]      = (_Float16)p0[r];
            ps[m * 32 + 16 + nl] = (_Float16)p1[r];
        }
        __syncthreads();

        // O += P (16x32) x V (32 j x 128 c), 8 channel chunks of 16
        v16h pa = frag_a16x32(ps, 32, 0);
#pragma unroll
        for (int t = 0; t < 8; ++t) {
            v16h bv = frag_b32x16(vt + (size_t)(t * 16) * 32, 32, 0);
            accO[t] = WMMA_F32_F16(pa, bv, accO[t]);
        }
    }

    // normalize and write out[b][c][i0+m]
#pragma unroll
    for (int t = 0; t < 8; ++t)
#pragma unroll
        for (int r = 0; r < 8; ++r) {
            int m = r + (hi << 3);
            int c = t * 16 + nl;
            float val = accO[t][r] / rowL[r];
            out[cb + (size_t)c * HW + i0 + m] = (_Float16)val;
        }
}

// ---------------------------------------------------------------------------
// Elementwise helpers
// ---------------------------------------------------------------------------
__global__ void cast_f32_to_f16_kernel(const float* __restrict__ in,
                                       _Float16* __restrict__ out, size_t n) {
    size_t i = (size_t)blockIdx.x * blockDim.x + threadIdx.x;
    if (i < n) out[i] = (_Float16)in[i];
}

__global__ void sft_apply_kernel(float* __restrict__ h,
                                 const float* __restrict__ gamma,
                                 const float* __restrict__ beta, size_t n) {
    size_t i = (size_t)blockIdx.x * blockDim.x + threadIdx.x;
    if (i < n) h[i] = fmaf(h[i], gamma[i], beta[i]);
}

// ---------------------------------------------------------------------------
// Host-side orchestration
// ---------------------------------------------------------------------------
extern "C" void kernel_launch(void* const* d_in, const int* in_sizes, int n_in,
                              void* d_out, int out_size, void* d_ws, size_t ws_size,
                              hipStream_t stream) {
    const int B = 4, C = 128, H = 64, W = 64, HW = H * W, GROUPS = 16;
    const int G = 64, GH = 256, GW = 256;
    const size_t XSZ = (size_t)B * C * HW;          // 2,097,152
    const size_t GSZ = (size_t)B * G * GH * GW;     // 16,777,216
    const size_t SSZ = (size_t)B * 128 * 128 * 128; // 8,388,608

    const float* x        = (const float*)d_in[0];
    const float* guidance = (const float*)d_in[1];
    const float* rb_gn1_g = (const float*)d_in[2];
    const float* rb_gn1_b = (const float*)d_in[3];
    const float* rb_a1    = (const float*)d_in[4];
    const float* rb_w1    = (const float*)d_in[5];
    const float* rb_b1    = (const float*)d_in[6];
    const float* rb_gn2_g = (const float*)d_in[7];
    const float* rb_gn2_b = (const float*)d_in[8];
    const float* rb_a2    = (const float*)d_in[9];
    const float* rb_w2    = (const float*)d_in[10];
    const float* rb_b2    = (const float*)d_in[11];
    const float* sft_sw   = (const float*)d_in[12];
    const float* sft_sb   = (const float*)d_in[13];
    const float* sft_gw   = (const float*)d_in[14];
    const float* sft_gb   = (const float*)d_in[15];
    const float* sft_bw   = (const float*)d_in[16];
    const float* sft_bb   = (const float*)d_in[17];
    const float* att_gn_g = (const float*)d_in[18];
    const float* att_gn_b = (const float*)d_in[19];
    const float* q_w      = (const float*)d_in[20];
    const float* q_b      = (const float*)d_in[21];
    const float* k_w      = (const float*)d_in[22];
    const float* k_b      = (const float*)d_in[23];
    const float* v_w      = (const float*)d_in[24];
    const float* v_b      = (const float*)d_in[25];
    const float* p_w      = (const float*)d_in[26];
    const float* p_b      = (const float*)d_in[27];
    float* out = (float*)d_out;

    char* cur = (char*)d_ws;
    auto alloc = [&](size_t bytes) -> char* {
        char* p = cur;
        cur += (bytes + 255) & ~(size_t)255;
        return p;
    };
    _Float16* t16a  = (_Float16*)alloc(XSZ * 2);
    float*    h1    = (float*)   alloc(XSZ * 4);
    _Float16* t16b  = (_Float16*)alloc(XSZ * 2);
    float*    hbuf  = (float*)   alloc(XSZ * 4);
    _Float16* g16   = (_Float16*)alloc(GSZ * 2);
    _Float16* s16   = (_Float16*)alloc(SSZ * 2);
    float*    gamma = (float*)   alloc(XSZ * 4);
    float*    beta  = (float*)   alloc(XSZ * 4);
    _Float16* hn16  = (_Float16*)alloc(XSZ * 2);
    _Float16* qT    = (_Float16*)alloc(XSZ * 2);   // token-major [b][hw][c]
    _Float16* kTt   = (_Float16*)alloc(XSZ * 2);   // token-major [b][hw][c]
    _Float16* v16   = (_Float16*)alloc(XSZ * 2);   // channel-major [b][c][hw]
    _Float16* ho16  = (_Float16*)alloc(XSZ * 2);
    (void)ws_size; (void)n_in; (void)in_sizes; (void)out_size;

    // 1) guidance f32 -> f16
    cast_f32_to_f16_kernel<<<(unsigned)((GSZ + 255) / 256), 256, 0, stream>>>(guidance, g16, GSZ);

    // 2) GN1 + PReLU
    gn_prelu_kernel<<<B * GROUPS, 256, 0, stream>>>(x, t16a, rb_gn1_g, rb_gn1_b, rb_a1, C, HW, GROUPS);

    // 3) conv1 3x3 (128->128)
    conv_wmma_kernel<<<dim3(HW / 128, C / 16, B), 128, 0, stream>>>(
        t16a, rb_w1, rb_b1, h1, nullptr, nullptr, nullptr,
        C, C, H, W, H, W, 3, 1, 1, 0);

    // 4) GN2 + PReLU
    gn_prelu_kernel<<<B * GROUPS, 256, 0, stream>>>(h1, t16b, rb_gn2_g, rb_gn2_b, rb_a2, C, HW, GROUPS);

    // 5) conv2 3x3 + residual (h = x + conv2)
    conv_wmma_kernel<<<dim3(HW / 128, C / 16, B), 128, 0, stream>>>(
        t16b, rb_w2, rb_b2, hbuf, nullptr, nullptr, x,
        C, C, H, W, H, W, 3, 1, 1, 0);

    // 6) SFT shared conv: guidance 64ch 256x256 -> 128ch 128x128, stride 2, leaky
    conv_wmma_kernel<<<dim3((128 * 128) / 128, 128 / 16, B), 128, 0, stream>>>(
        g16, sft_sw, sft_sb, nullptr, s16, nullptr, nullptr,
        G, 128, GH, GW, 128, 128, 3, 2, 1, 1);

    // 7,8) gamma / beta convs: 128ch 128x128 -> 128ch 64x64, stride 2
    conv_wmma_kernel<<<dim3(HW / 128, C / 16, B), 128, 0, stream>>>(
        s16, sft_gw, sft_gb, gamma, nullptr, nullptr, nullptr,
        128, C, 128, 128, H, W, 3, 2, 1, 0);
    conv_wmma_kernel<<<dim3(HW / 128, C / 16, B), 128, 0, stream>>>(
        s16, sft_bw, sft_bb, beta, nullptr, nullptr, nullptr,
        128, C, 128, 128, H, W, 3, 2, 1, 0);

    // 9) h = h * gamma + beta (in place)
    sft_apply_kernel<<<(unsigned)((XSZ + 255) / 256), 256, 0, stream>>>(hbuf, gamma, beta, XSZ);

    // 10) attention GN (no PReLU)
    gn_prelu_kernel<<<B * GROUPS, 256, 0, stream>>>(hbuf, hn16, att_gn_g, att_gn_b, nullptr, C, HW, GROUPS);

    // 11) q/k 1x1 convs -> token-major f16; v -> channel-major f16
    conv_wmma_kernel<<<dim3(HW / 128, C / 16, B), 128, 0, stream>>>(
        hn16, q_w, q_b, nullptr, nullptr, qT, nullptr, C, C, H, W, H, W, 1, 1, 0, 0);
    conv_wmma_kernel<<<dim3(HW / 128, C / 16, B), 128, 0, stream>>>(
        hn16, k_w, k_b, nullptr, nullptr, kTt, nullptr, C, C, H, W, H, W, 1, 1, 0, 0);
    conv_wmma_kernel<<<dim3(HW / 128, C / 16, B), 128, 0, stream>>>(
        hn16, v_w, v_b, nullptr, v16, nullptr, nullptr, C, C, H, W, H, W, 1, 1, 0, 0);

    // 12) flash attention (TDM-staged tiles)
    attn_flash_kernel<<<dim3(HW / 16, B), 32, 0, stream>>>(
        qT, kTt, v16, ho16, C, HW, 0.08838834764831845f /* 128^-0.5 */);

    // 13) projection 1x1 conv + final residual -> d_out
    conv_wmma_kernel<<<dim3(HW / 128, C / 16, B), 128, 0, stream>>>(
        ho16, p_w, p_b, out, nullptr, nullptr, hbuf, C, C, H, W, H, W, 1, 1, 0, 0);
}
